// HybridGhostGNN_40450001994225
// MI455X (gfx1250) — matrix operand order repaired and checked by reading, not compiled, not run on hardware
//
#include <hip/hip_runtime.h>
#include <hip/hip_bf16.h>
#include <math.h>

// ---------------- types for WMMA ----------------
typedef __attribute__((ext_vector_type(16))) __bf16 v16bf;
typedef __attribute__((ext_vector_type(8)))  float  v8f;

#define HID 128

// ---------------- helpers ----------------
__device__ __forceinline__ unsigned int f2bf(float f) {
    unsigned int u = __float_as_uint(f);
    u += 0x7FFFu + ((u >> 16) & 1u);   // round-to-nearest-even
    return u >> 16;
}
__device__ __forceinline__ unsigned int packbf(float lo, float hi) {
    return f2bf(lo) | (f2bf(hi) << 16);
}

// ---------------- generic fill ----------------
__global__ void fill_f32(float* __restrict__ p, float v, long long n) {
    long long i = (long long)blockIdx.x * blockDim.x + threadIdx.x;
    if (i < n) p[i] = v;
}

// ---------------- degree ----------------
__global__ void deg_scatter(const long long* __restrict__ dst, float* __restrict__ deg, int nE) {
    int e = blockIdx.x * blockDim.x + threadIdx.x;
    if (e < nE) atomicAdd(&deg[(int)dst[e]], 1.0f);
}
__global__ void deg_invert(float* __restrict__ d, int n) {
    int i = blockIdx.x * blockDim.x + threadIdx.x;
    if (i < n) d[i] = 1.0f / fmaxf(d[i], 1.0f);
}

// ---------------- layer 0 aggregation (6-dim) ----------------
__global__ void scatter6(const float* __restrict__ x, const long long* __restrict__ src,
                         const long long* __restrict__ dst, float* __restrict__ agg6, int nE) {
    int idx = blockIdx.x * blockDim.x + threadIdx.x;
    if (idx >= nE * 6) return;
    int e = idx / 6, j = idx - e * 6;
    int s = (int)src[e], d = (int)dst[e];
    atomicAdd(&agg6[d * 6 + j], x[s * 6 + j]);
}

// ---------------- layer 0 forward: 6 -> 128 (scalar, tiny K) ----------------
__global__ __launch_bounds__(128) void l0_forward(
    const float* __restrict__ x, const float* __restrict__ agg6, const float* __restrict__ invd,
    const float* __restrict__ wl, const float* __restrict__ bl, const float* __restrict__ wr,
    float* __restrict__ h, int nNodes)
{
    __shared__ float sx[8][6], sm[8][6];
    int nb = blockIdx.x * 8;
    int tid = threadIdx.x;
    if (tid < 48) {
        int node = nb + tid / 6, j = tid % 6;
        float xv = 0.f, mv = 0.f;
        if (node < nNodes) {
            xv = x[node * 6 + j];
            mv = agg6[node * 6 + j] * invd[node];
        }
        sx[tid / 6][j] = xv;
        sm[tid / 6][j] = mv;
    }
    __syncthreads();
    float wlr[6], wrr[6];
#pragma unroll
    for (int j = 0; j < 6; ++j) { wlr[j] = wl[tid * 6 + j]; wrr[j] = wr[tid * 6 + j]; }
    float bb = bl[tid];
    for (int i = 0; i < 8; ++i) {
        int node = nb + i;
        if (node >= nNodes) break;
        float acc = bb;
#pragma unroll
        for (int j = 0; j < 6; ++j) acc += wlr[j] * sm[i][j] + wrr[j] * sx[i][j];
        h[node * HID + tid] = acc;
    }
}

// ---------------- batch-norm statistics ----------------
__global__ __launch_bounds__(128) void bn_stats(const float* __restrict__ h, float* __restrict__ stats,
                                                int nRows, int rowsPerBlock) {
    int f = threadIdx.x;
    int r0 = blockIdx.x * rowsPerBlock;
    int r1 = min(r0 + rowsPerBlock, nRows);
    float s = 0.f, q = 0.f;
    for (int r = r0; r < r1; ++r) {
        float v = h[r * HID + f];
        s += v; q += v * v;
    }
    atomicAdd(&stats[f], s);
    atomicAdd(&stats[HID + f], q);
}
__global__ void bn_final(const float* __restrict__ stats, const float* __restrict__ gamma,
                         const float* __restrict__ beta, float* __restrict__ coef, int nRows) {
    int f = threadIdx.x;   // 128 threads
    float inv_n = 1.0f / (float)nRows;
    float mu = stats[f] * inv_n;
    float var = stats[HID + f] * inv_n - mu * mu;
    float rs = rsqrtf(var + 1e-5f) * gamma[f];
    coef[f] = rs;
    coef[HID + f] = beta[f] - mu * rs;
}
__global__ void norm_relu(float* __restrict__ h, const float* __restrict__ coef, long long total) {
    long long i = (long long)blockIdx.x * blockDim.x + threadIdx.x;
    if (i >= total) return;
    int f = (int)(i & (HID - 1));
    float v = h[i] * coef[f] + coef[HID + f];
    h[i] = v > 0.f ? v : 0.f;
}

// ---------------- 128-dim aggregation scatter: one wave per edge ----------------
// Lane loads a coalesced float4 of the source row (global_load_b128) and issues
// 4 f32 atomics into the destination row. Edge indices are wave-uniform.
__global__ __launch_bounds__(256) void scatter128(const float* __restrict__ h,
                                                  const long long* __restrict__ src,
                                                  const long long* __restrict__ dst,
                                                  float* __restrict__ agg, int nE) {
    int wave = threadIdx.x >> 5, lane = threadIdx.x & 31;
    int e = blockIdx.x * 8 + wave;
    if (e >= nE) return;
    int s = (int)src[e], d = (int)dst[e];
    const float4* hs = (const float4*)(h + (size_t)s * HID);
    float4 v = hs[lane];
    float* ad = agg + (size_t)d * HID + lane * 4;
    atomicAdd(ad + 0, v.x);
    atomicAdd(ad + 1, v.y);
    atomicAdd(ad + 2, v.z);
    atomicAdd(ad + 3, v.w);
}

// ---------------- pack activations to bf16 K-pairs: [mean_nb | h] (K=256) ----------------
__global__ void pack_act(const float* __restrict__ agg, const float* __restrict__ invd,
                         const float* __restrict__ h, unsigned int* __restrict__ Apk, int nNodes) {
    int idx = blockIdx.x * blockDim.x + threadIdx.x;
    if (idx >= nNodes * HID) return;
    int node = idx >> 7, kp = idx & (HID - 1);
    float a, b;
    if (kp < 64) {
        float s = invd[node];
        a = agg[node * HID + 2 * kp] * s;
        b = agg[node * HID + 2 * kp + 1] * s;
    } else {
        int k = (kp - 64) * 2;
        a = h[node * HID + k];
        b = h[node * HID + k + 1];
    }
    Apk[idx] = packbf(a, b);
}

// ---------------- pack weights: B[k][n], k<128 from w_l1, else w_r1 ----------------
__global__ void pack_w1(const float* __restrict__ wl, const float* __restrict__ wr,
                        unsigned int* __restrict__ Wpk) {
    int idx = blockIdx.x * blockDim.x + threadIdx.x;   // 128 kp x 128 n
    if (idx >= HID * HID) return;
    int kp = idx >> 7, n = idx & (HID - 1);
    float a, b;
    if (kp < 64) { int k = 2 * kp;          a = wl[n * HID + k]; b = wl[n * HID + k + 1]; }
    else         { int k = 2 * (kp - 64);   a = wr[n * HID + k]; b = wr[n * HID + k + 1]; }
    Wpk[idx] = packbf(a, b);
}

// ---------------- layer-1 GEMM: [N x 256]bf16 @ [256 x 128]bf16 -> f32, WMMA ----------------
// Block = 256 threads = 8 waves. Tile: 64 rows x 128 cols.
// A-tile staged into LDS with async global->LDS DMA (ASYNCcnt path, bypasses VGPRs).
__global__ __launch_bounds__(256) void gemm_wmma_l1(
    const unsigned int* __restrict__ Apk,   // [N][128] bf16 pairs (K=256)
    const unsigned int* __restrict__ Wpk,   // [128][128] bf16 pairs (kp-major, n)
    const float* __restrict__ bias,         // [128]
    float* __restrict__ out,                // [N][128]
    int nRows)
{
    __shared__ unsigned int sA[64 * HID];   // 32 KB
    const int tid = threadIdx.x;
    const int rowBase = blockIdx.x * 64;

    // async stage of the A tile: 64 rows x 512B, in 16B chunks (2048 chunks, 8/thread)
    for (int i = tid; i < 64 * 32; i += 256) {
        int r = i >> 5;               // row within tile
        int c16 = i & 31;             // 16B chunk within row
        int gr = rowBase + r;
        if (gr >= nRows) gr = nRows - 1;   // clamp; stores guarded later
        unsigned long long ga =
            (unsigned long long)(const char*)(Apk + (size_t)gr * HID) + (size_t)c16 * 16;
        unsigned int lds = (unsigned int)(size_t)(&sA[0]) + (unsigned int)i * 16u;
        asm volatile("global_load_async_to_lds_b128 %0, %1, off"
                     :: "v"(lds), "v"(ga) : "memory");
    }
    asm volatile("s_wait_asynccnt 0" ::: "memory");
    __syncthreads();

    const int lane = tid & 31;
    const int wave = tid >> 5;
    const int wr   = wave & 3;
    const int wc   = wave >> 2;
    const int half = lane >> 4;
    const int lm   = lane & 15;

    union Frag { v16bf v; unsigned int u[8]; };
    v8f zero = {};
    v8f acc[4];
#pragma unroll
    for (int t = 0; t < 4; ++t) acc[t] = zero;

    const int rowA = wr * 16 + lm;
#pragma unroll
    for (int kk = 0; kk < 8; ++kk) {           // K = 256 in steps of 32
        Frag a;
#pragma unroll
        for (int v = 0; v < 8; ++v) {
            // A 16-bit layout: kpLocal = (v>>2)*8 + half*4 + (v&3)
            int kp = kk * 16 + ((v >> 2) * 8 + half * 4 + (v & 3));
            a.u[v] = sA[rowA * HID + kp];
        }
#pragma unroll
        for (int t = 0; t < 4; ++t) {
            Frag b;
            int n = wc * 64 + t * 16 + lm;
#pragma unroll
            for (int v = 0; v < 8; ++v) {
                // B 16-bit layout: kpLocal = (v>>2)*8 + half*4 + (v&3)
                int kp = kk * 16 + ((v >> 2) * 8 + half * 4 + (v & 3));
                b.u[v] = Wpk[kp * HID + n];
            }
            acc[t] = __builtin_amdgcn_wmma_f32_16x16x32_bf16(
                false, a.v, false, b.v, (short)0, acc[t], false, false);
        }
    }

    // store D: lanes 0-15 -> M=r, lanes 16-31 -> M=8+r; N = lane&15
#pragma unroll
    for (int t = 0; t < 4; ++t) {
        int n = wc * 64 + t * 16 + lm;
        float bv = bias[n];
#pragma unroll
        for (int r = 0; r < 8; ++r) {
            int row = rowBase + wr * 16 + half * 8 + r;
            if (row < nRows) out[row * HID + n] = acc[t][r] + bv;
        }
    }
}

// ---------------- layer 2 head: 128 -> 1 + sigmoid (wave per node) ----------------
__global__ __launch_bounds__(256) void final_l2(
    const float* __restrict__ agg, const float* __restrict__ invd, const float* __restrict__ h,
    const float* __restrict__ wl2, const float* __restrict__ bl2, const float* __restrict__ wr2,
    float* __restrict__ out, int nNodes)
{
    int wave = threadIdx.x >> 5, lane = threadIdx.x & 31;
    int node = blockIdx.x * 8 + wave;
    if (node >= nNodes) return;
    float s = invd[node];
    float acc = 0.f;
#pragma unroll
    for (int i = 0; i < 4; ++i) {
        int f = lane + i * 32;
        acc += agg[node * HID + f] * s * wl2[f] + h[node * HID + f] * wr2[f];
    }
#pragma unroll
    for (int off = 16; off >= 1; off >>= 1) acc += __shfl_xor(acc, off, 32);
    if (lane == 0) {
        float z = acc + bl2[0];
        out[node] = 1.0f / (1.0f + expf(-z));
    }
}

// ---------------- launch ----------------
extern "C" void kernel_launch(void* const* d_in, const int* in_sizes, int n_in,
                              void* d_out, int out_size, void* d_ws, size_t ws_size,
                              hipStream_t stream) {
    const float*     x    = (const float*)d_in[0];
    const long long* edge = (const long long*)d_in[1];
    const float* w_l0 = (const float*)d_in[2];
    const float* b_l0 = (const float*)d_in[3];
    const float* w_r0 = (const float*)d_in[4];
    const float* bn_g0 = (const float*)d_in[5];
    const float* bn_b0 = (const float*)d_in[6];
    const float* w_l1 = (const float*)d_in[7];
    const float* b_l1 = (const float*)d_in[8];
    const float* w_r1 = (const float*)d_in[9];
    const float* bn_g1 = (const float*)d_in[10];
    const float* bn_b1 = (const float*)d_in[11];
    const float* w_l2 = (const float*)d_in[12];
    const float* b_l2 = (const float*)d_in[13];
    const float* w_r2 = (const float*)d_in[14];

    const int nNodes = in_sizes[0] / 6;
    const int nEdges = in_sizes[1] / 2;
    const long long* src = edge;
    const long long* dst = edge + nEdges;

    // workspace carve-up
    char* w = (char*)d_ws;
    float* invd = (float*)w;                               w += ((size_t)nNodes * 4 + 511) & ~511ull;
    float* agg  = (float*)w;                               w += (size_t)nNodes * HID * 4;
    float* hbuf = (float*)w;                               w += (size_t)nNodes * HID * 4;
    unsigned int* Apk = (unsigned int*)w;                  w += (size_t)nNodes * HID * 4;
    unsigned int* Wpk = (unsigned int*)w;                  w += (size_t)HID * HID * 4;
    float* stats = (float*)w;                              w += 256 * 4;
    float* coef  = (float*)w;

    const int T = 256;
    const long long nh = (long long)nNodes * HID;
    dim3 blk(T);

    // degrees
    fill_f32<<<(nNodes + T - 1) / T, blk, 0, stream>>>(invd, 0.f, nNodes);
    deg_scatter<<<(nEdges + T - 1) / T, blk, 0, stream>>>(dst, invd, nEdges);
    deg_invert<<<(nNodes + T - 1) / T, blk, 0, stream>>>(invd, nNodes);

    // pack layer-1 weights (any time before GEMM)
    pack_w1<<<(HID * HID + T - 1) / T, blk, 0, stream>>>(w_l1, w_r1, Wpk);

    // ---- layer 0 ----
    fill_f32<<<((long long)nNodes * 6 + T - 1) / T, blk, 0, stream>>>(agg, 0.f, (long long)nNodes * 6);
    scatter6<<<(nEdges * 6 + T - 1) / T, blk, 0, stream>>>(x, src, dst, agg, nEdges);
    l0_forward<<<(nNodes + 7) / 8, dim3(128), 0, stream>>>(x, agg, invd, w_l0, b_l0, w_r0, hbuf, nNodes);

    fill_f32<<<1, blk, 0, stream>>>(stats, 0.f, 256);
    bn_stats<<<(nNodes + 511) / 512, dim3(128), 0, stream>>>(hbuf, stats, nNodes, 512);
    bn_final<<<1, dim3(128), 0, stream>>>(stats, bn_g0, bn_b0, coef, nNodes);
    norm_relu<<<(nh + T - 1) / T, blk, 0, stream>>>(hbuf, coef, nh);

    // ---- layer 1 ----
    fill_f32<<<(nh + T - 1) / T, blk, 0, stream>>>(agg, 0.f, nh);
    scatter128<<<(nEdges + 7) / 8, blk, 0, stream>>>(hbuf, src, dst, agg, nEdges);
    pack_act<<<(nh + T - 1) / T, blk, 0, stream>>>(agg, invd, hbuf, Apk, nNodes);
    gemm_wmma_l1<<<(nNodes + 63) / 64, blk, 0, stream>>>(Apk, Wpk, b_l1, hbuf, nNodes);

    fill_f32<<<1, blk, 0, stream>>>(stats, 0.f, 256);
    bn_stats<<<(nNodes + 511) / 512, dim3(128), 0, stream>>>(hbuf, stats, nNodes, 512);
    bn_final<<<1, dim3(128), 0, stream>>>(stats, bn_g1, bn_b1, coef, nNodes);
    norm_relu<<<(nh + T - 1) / T, blk, 0, stream>>>(hbuf, coef, nh);

    // ---- layer 2 ----
    fill_f32<<<(nh + T - 1) / T, blk, 0, stream>>>(agg, 0.f, nh);
    scatter128<<<(nEdges + 7) / 8, blk, 0, stream>>>(hbuf, src, dst, agg, nEdges);
    final_l2<<<(nNodes + 7) / 8, blk, 0, stream>>>(agg, invd, hbuf, w_l2, b_l2, w_r2,
                                                   (float*)d_out, nNodes);
}
